// GroupedDilatedAttention_LocalGKV_18588618457408
// MI455X (gfx1250) — compile-verified, matrix operand order checked by reading, not run on hardware
//
#include <hip/hip_runtime.h>
#include <hip/hip_bf16.h>

typedef __attribute__((ext_vector_type(16))) __bf16 v16bf;
typedef __attribute__((ext_vector_type(8)))  float  v8f;

// Load 16 f32 (8 consecutive at p, 8 consecutive at p+16) and convert to bf16.
// With lane-half offset +8, this realizes the CDNA5 16-bit A/B operand layout:
// lanes 0-15 hold K=0..7 / 16..23, lanes 16-31 hold K=8..15 / 24..31.
__device__ __forceinline__ v16bf load16_bf16(const float* __restrict__ p) {
    float f[16];
    *(float4*)(f + 0)  = ((const float4*)p)[0];
    *(float4*)(f + 4)  = ((const float4*)p)[1];
    *(float4*)(f + 8)  = ((const float4*)(p + 16))[0];
    *(float4*)(f + 12) = ((const float4*)(p + 16))[1];
    v16bf r;
#pragma unroll
    for (int e = 0; e < 16; ++e) r[e] = (__bf16)f[e];
    return r;
}

#define WMMA_BF16(A, B, C) \
    __builtin_amdgcn_wmma_f32_16x16x32_bf16(false, (A), false, (B), (short)0, (C), false, false)

// ---------------------------------------------------------------------------
// Kernel 1: QKV projection. 16 GEMMs (g in 0..3, b2 in 0..3):
//   qkv[g][b2*576 + o][l] = sum_c w[o,c] * x[(g*4096+l)*768 + b2*192 + c]
// M=576, N=4096, K=192. Wave tile 32x64: 2 A-subtiles x 4 B-subtiles ->
// 8 accumulators; 6 tile-loads feed 8 WMMAs per K-step of 32.
// ---------------------------------------------------------------------------
__global__ __launch_bounds__(256) void qkv_gemm(const float* __restrict__ x,
                                                const float* __restrict__ w,
                                                float* __restrict__ qkvb) {
    const int lane = threadIdx.x & 31;
    const int wid  = blockIdx.x * 8 + (threadIdx.x >> 5);
    if (wid >= 16 * 18 * 64) return;           // uniform per-wave: EXEC stays full
    const int nt = wid & 63;                   // 64-wide N supertile
    const int mt = (wid >> 6) % 18;            // 32-wide M supertile
    const int gb = wid / (18 * 64);
    const int b2 = gb & 3;
    const int g  = gb >> 2;

    const int half = lane >> 4;                // K-half select per CDNA5 16-bit layout
    const int lr   = lane & 15;
    const int m0   = mt * 32 + lr;             // A rows m0, m0+16
    const int n0   = nt * 64 + lr;             // B cols n0 + 16*j

    const float* ap = w + (size_t)m0 * 192 + half * 8;
    const float* bp = x + ((size_t)(g * 4096 + n0)) * 768 + b2 * 192 + half * 8;

    v8f acc[2][4] = {};
#pragma unroll
    for (int kb = 0; kb < 192; kb += 32) {
        const v16bf a0 = load16_bf16(ap + kb);
        const v16bf a1 = load16_bf16(ap + (size_t)16 * 192 + kb);
#pragma unroll
        for (int j = 0; j < 4; ++j) {
            const v16bf b = load16_bf16(bp + (size_t)j * 16 * 768 + kb);
            acc[0][j] = WMMA_BF16(a0, b, acc[0][j]);
            acc[1][j] = WMMA_BF16(a1, b, acc[1][j]);
        }
    }
    // D layout: VGPR r -> M = r + half*8, N = lr
    const int rowbase = g * 2304 + b2 * 576 + mt * 32 + half * 8;
    const int colbase = nt * 64 + lr;
#pragma unroll
    for (int mi = 0; mi < 2; ++mi)
#pragma unroll
        for (int j = 0; j < 4; ++j) {
            float* op = qkvb + (size_t)(rowbase + mi * 16) * 4096 + colbase + j * 16;
#pragma unroll
            for (int r = 0; r < 8; ++r) op[(size_t)r * 4096] = acc[mi][j][r];
        }
}

// ---------------------------------------------------------------------------
// Kernel 2: dilated 3x3 local attention, one thread per (g,i,b2,l).
// q/k/v rows (torch-faithful): base = g*2304 + i*768 + {0,256,512} + b2*64.
// Zero-padded unfold: OOB tap -> logit exactly 0 (kept in softmax), v = 0.
// Output written pre-scrambled so proj sees row-major y[16384, 768].
// ---------------------------------------------------------------------------
__global__ __launch_bounds__(256) void dilated_attn(const float* __restrict__ qkvb,
                                                    float* __restrict__ y) {
    const int tid = blockIdx.x * 256 + threadIdx.x;   // = chunk*4096 + l
    const int l   = tid & 4095;
    int r = tid >> 12;                                 // 0..47
    const int b2 = r & 3;  r >>= 2;
    const int i  = r % 3;
    const int g  = r / 3;
    const int dil = i + 1;                             // DILATIONS = (1,2,3)
    const int h = l >> 6, w = l & 63;

    const int rowq = g * 2304 + i * 768 + b2 * 64;
    const int rowk = rowq + 256;
    const int rowv = rowq + 512;

    int   lofs[9];
    float vmask[9];
#pragma unroll
    for (int t9 = 0; t9 < 9; ++t9) {
        const int hh = h + (t9 / 3 - 1) * dil;
        const int ww = w + (t9 % 3 - 1) * dil;
        const bool ok = ((unsigned)hh < 64u) && ((unsigned)ww < 64u);
        vmask[t9] = ok ? 1.0f : 0.0f;
        lofs[t9]  = ok ? (hh * 64 + ww) : l;           // safe in-range address
    }

    // Pass 1: nine QK dot products over 64 channels
    float acc[9];
#pragma unroll
    for (int t9 = 0; t9 < 9; ++t9) acc[t9] = 0.0f;
    const float* qp = qkvb + (size_t)rowq * 4096 + l;
    const float* kp = qkvb + (size_t)rowk * 4096;
    for (int c = 0; c < 64; ++c) {
        const float qc = qp[(size_t)c * 4096];
        const float* kc = kp + (size_t)c * 4096;
#pragma unroll
        for (int t9 = 0; t9 < 9; ++t9) acc[t9] += qc * kc[lofs[t9]];
    }

    // Softmax over 9 taps (scale = 64^-0.5 = 0.125); OOB logits are exactly 0.
    float logit[9];
    float mx = -3.0e38f;
#pragma unroll
    for (int t9 = 0; t9 < 9; ++t9) {
        logit[t9] = acc[t9] * 0.125f * vmask[t9];
        mx = fmaxf(mx, logit[t9]);
    }
    float s = 0.0f, p[9];
#pragma unroll
    for (int t9 = 0; t9 < 9; ++t9) { p[t9] = __expf(logit[t9] - mx); s += p[t9]; }
    const float inv = 1.0f / s;
#pragma unroll
    for (int t9 = 0; t9 < 9; ++t9) p[t9] = p[t9] * inv * vmask[t9];  // zero OOB values

    // Pass 2: weighted sum of V, store scrambled: y[tid*64 + c]
    const float* vp = qkvb + (size_t)rowv * 4096;
    float* yo = y + (size_t)tid * 64;
    for (int c = 0; c < 64; ++c) {
        const float* vc = vp + (size_t)c * 4096;
        float o = 0.0f;
#pragma unroll
        for (int t9 = 0; t9 < 9; ++t9) o += p[t9] * vc[lofs[t9]];
        yo[c] = o;
    }
}

// ---------------------------------------------------------------------------
// Kernel 3: output projection. out[m, j] = sum_k y[m,k] * pw[j,k] + pb[j]
// M=16384, N=768, K=768. Wave tile 32x64 (8 accumulators), 24 K-steps.
// ---------------------------------------------------------------------------
__global__ __launch_bounds__(256) void proj_gemm(const float* __restrict__ y,
                                                 const float* __restrict__ pw,
                                                 const float* __restrict__ pb,
                                                 float* __restrict__ out) {
    const int lane = threadIdx.x & 31;
    const int wid  = blockIdx.x * 8 + (threadIdx.x >> 5);
    if (wid >= 512 * 12) return;
    const int nt = wid % 12;                   // 64-wide N supertile
    const int mt = wid / 12;                   // 32-wide M supertile

    const int half = lane >> 4;
    const int lr   = lane & 15;
    const int m0   = mt * 32 + lr;
    const int col0 = nt * 64 + lr;

    const float* ap = y  + (size_t)m0 * 768 + half * 8;
    const float* bp = pw + (size_t)col0 * 768 + half * 8;

    v8f acc[2][4] = {};
#pragma unroll 2
    for (int kb = 0; kb < 768; kb += 32) {
        const v16bf a0 = load16_bf16(ap + kb);
        const v16bf a1 = load16_bf16(ap + (size_t)16 * 768 + kb);
#pragma unroll
        for (int j = 0; j < 4; ++j) {
            const v16bf b = load16_bf16(bp + (size_t)j * 16 * 768 + kb);
            acc[0][j] = WMMA_BF16(a0, b, acc[0][j]);
            acc[1][j] = WMMA_BF16(a1, b, acc[1][j]);
        }
    }
    const int rowbase = mt * 32 + half * 8;
#pragma unroll
    for (int mi = 0; mi < 2; ++mi)
#pragma unroll
        for (int j = 0; j < 4; ++j) {
            const int col = col0 + j * 16;
            const float bias = pb[col];
            float* op = out + (size_t)(rowbase + mi * 16) * 768 + col;
#pragma unroll
            for (int r = 0; r < 8; ++r) op[(size_t)r * 768] = acc[mi][j][r] + bias;
        }
}

// ---------------------------------------------------------------------------
// Launch: inputs are x[4,64,64,768] f32, qkv_w[576,192], proj_w[768,768],
// proj_b[768]. Workspace: qkv buffer (4*2304*4096 f32 = 151 MB) + scrambled
// y (16384*768 f32 = 50 MB) = 192 MiB of d_ws. Output f32 [4,64,64,768].
// ---------------------------------------------------------------------------
extern "C" void kernel_launch(void* const* d_in, const int* in_sizes, int n_in,
                              void* d_out, int out_size, void* d_ws, size_t ws_size,
                              hipStream_t stream) {
    const float* x    = (const float*)d_in[0];
    const float* qkvw = (const float*)d_in[1];
    const float* pw   = (const float*)d_in[2];
    const float* pb   = (const float*)d_in[3];
    float* out  = (float*)d_out;

    float* qkvb = (float*)d_ws;
    float* ybuf = qkvb + (size_t)4 * 2304 * 4096;

    // 16 gb-pairs * 18 M-supertiles * 64 N-supertiles = 18432 waves / 8 per block
    qkv_gemm<<<2304, 256, 0, stream>>>(x, qkvw, qkvb);
    // 196608 positions / 256 threads
    dilated_attn<<<768, 256, 0, stream>>>(qkvb, ybuf);
    // 512 M-supertiles * 12 N-supertiles = 6144 waves / 8 per block
    proj_gemm<<<768, 256, 0, stream>>>(ybuf, pw, pb, out);
}
